// GAT_Block_18949395710591
// MI455X (gfx1250) — matrix operand order repaired and checked by reading, not compiled, not run on hardware
//
#include <hip/hip_runtime.h>
#include <hip/hip_bf16.h>

#define ALPHA 0.2f
#define EPS   1e-5f
#define B_    4
#define N_    1024
#define F_    256
#define H_    8
#define HID_  32
#define NF_   (N_ * F_)      // 262144
#define BNF_  (B_ * NF_)     // 1048576
#define BNH_  (B_ * N_ * H_) // 32768

typedef __attribute__((ext_vector_type(16))) __bf16 v16bf;
typedef __attribute__((ext_vector_type(8)))  float  v8f;
typedef __attribute__((ext_vector_type(2)))  float  v2f;

#if __has_builtin(__builtin_amdgcn_wmma_f32_16x16x4_f32)
#define HAVE_WMMA_F32 1
#else
#define HAVE_WMMA_F32 0
#endif

__device__ __forceinline__ unsigned lds_off(const void* p) {
  // addrspace(3) generic pointers carry the LDS byte offset in the low 32 bits
  return (unsigned)(unsigned long long)p;
}

// ---------------------------------------------------------------------------
// Dense feature GEMM:  C[m,n] = sum_k A[m,k] * Wt[n,k]
// A = h flattened [4096 x 256], Wt = torch Linear weight [256 x 256].
// Block = 8 waves sharing one 16-row A stripe (async-staged into LDS);
// each wave owns one 16x16 output tile (f32 WMMA, exact fp32 path).
// ---------------------------------------------------------------------------
__global__ __launch_bounds__(256)
void gemm_wmma_kernel(const float* __restrict__ A,
                      const float* __restrict__ Wt,
                      float* __restrict__ C) {
  const int K = F_, Nout = F_;
  const int tid  = threadIdx.x;
  const int lane = tid & 31;
  const int wave = tid >> 5;
  const int mt = blockIdx.x >> 1;               // all 8 waves share this M-tile
  const int nt = (blockIdx.x & 1) * 8 + wave;
  const int half = lane >> 4, l15 = lane & 15;
  const int col = nt * 16 + l15;                // B column for this lane

  __shared__ float ldsA[16 * 256];              // 16 KB A stripe [row][k]
  {
    const float* asrc = A + (size_t)mt * 16 * K;   // stripe is contiguous
    unsigned lbase = lds_off(&ldsA[0]);
#pragma unroll
    for (int q = 0; q < 4; q++) {
      unsigned boff = (unsigned)(tid + 256 * q) * 16u;   // 256 thr x 4 x 16B
      asm volatile("global_load_async_to_lds_b128 %0, %1, %2"
                   :: "v"(lbase + boff), "v"(boff), "s"(asrc) : "memory");
    }
  }
  asm volatile("s_wait_asynccnt 0" ::: "memory");
  __syncthreads();

  v8f acc = {};
#if HAVE_WMMA_F32
  for (int k0 = 0; k0 < K; k0 += 4) {
    // f32 A 16x4: VGPR v holds K = half*2 + v ; f32 B 4x16 symmetric.
    v2f a, b;
    a.x = ldsA[l15 * K + k0 + half * 2 + 0];
    a.y = ldsA[l15 * K + k0 + half * 2 + 1];
    b.x = Wt[col * K + k0 + half * 2 + 0];
    b.y = Wt[col * K + k0 + half * 2 + 1];
    acc = __builtin_amdgcn_wmma_f32_16x16x4_f32(false, a, false, b,
                                                (short)0, acc, false, false);
  }
#else
  for (int k0 = 0; k0 < K; k0 += 32) {
    v16bf a, b;
#pragma unroll
    for (int s = 0; s < 16; s++) {
      int v = s >> 1, hb = s & 1;
      int kq = (v < 4) ? (half * 8 + v * 2 + hb)
                       : (16 + half * 8 + (v - 4) * 2 + hb);
      a[s] = (__bf16)ldsA[l15 * K + k0 + kq];
      b[s] = (__bf16)Wt[col * K + k0 + half * 16 + s];
    }
    acc = __builtin_amdgcn_wmma_f32_16x16x32_bf16(false, a, false, b,
                                                  (short)0, acc, false, false);
  }
#endif
#pragma unroll
  for (int v = 0; v < 8; v++) {
    int r = mt * 16 + v + half * 8;             // C/D layout: M = vgpr + 8*half
    C[r * Nout + nt * 16 + l15] = acc[v];
  }
}

// ---------------------------------------------------------------------------
// src[b,n,h] = g[b,n,h,:] . a[:32] ;  dst = g . a[32:]
// ---------------------------------------------------------------------------
__global__ __launch_bounds__(256)
void src_dst_kernel(const float* __restrict__ g, const float* __restrict__ a,
                    float* __restrict__ src, float* __restrict__ dst) {
  int idx = blockIdx.x * 256 + threadIdx.x;
  if (idx >= BNH_) return;
  const float* gp = g + (size_t)idx * HID_;
  float s = 0.f, d = 0.f;
#pragma unroll
  for (int k = 0; k < HID_; k++) {
    s += gp[k] * a[k];
    d += gp[k] * a[HID_ + k];
  }
  src[idx] = s;
  dst[idx] = d;
}

// ---------------------------------------------------------------------------
// Softmax is over i (dim 1): per (b,j,h) compute  m_j = max_i e~_ij  and
// denom_j = sum_i exp(e~_ij - m_j), with e~ = leakyrelu(src_i + dst_j),
// masked by adj[i,j]. One block per (b,j); online max/sum + LDS tree reduce.
// ---------------------------------------------------------------------------
__global__ __launch_bounds__(256)
void col_stats_kernel(const float* __restrict__ src, const float* __restrict__ dst,
                      const int* __restrict__ adj,
                      float* __restrict__ mmax, float* __restrict__ denom) {
  const int b = blockIdx.x / N_;
  const int j = blockIdx.x % N_;
  const int tid = threadIdx.x;
  const float NINF = -__builtin_inff();

  float dv[H_], m[H_], s[H_];
#pragma unroll
  for (int h = 0; h < H_; h++) {
    dv[h] = dst[(b * N_ + j) * H_ + h];
    m[h] = NINF;
    s[h] = 0.f;
  }
  for (int i = tid; i < N_; i += 256) {
    if (adj[i * N_ + j]) {
      const float* sp = src + (size_t)(b * N_ + i) * H_;
#pragma unroll
      for (int h = 0; h < H_; h++) {
        float u = sp[h] + dv[h];
        float e = (u >= 0.f) ? u : ALPHA * u;
        if (e > m[h]) {
          s[h] = s[h] * __expf(m[h] - e) + 1.f;
          m[h] = e;
        } else {
          s[h] += __expf(e - m[h]);
        }
      }
    }
  }
  __shared__ float sm[256 * H_];
  __shared__ float ss[256 * H_];
#pragma unroll
  for (int h = 0; h < H_; h++) { sm[tid * H_ + h] = m[h]; ss[tid * H_ + h] = s[h]; }
  __syncthreads();
  for (int off = 128; off > 0; off >>= 1) {
    if (tid < off) {
#pragma unroll
      for (int h = 0; h < H_; h++) {
        float m1 = sm[tid * H_ + h],         s1 = ss[tid * H_ + h];
        float m2 = sm[(tid + off) * H_ + h], s2 = ss[(tid + off) * H_ + h];
        float M = fmaxf(m1, m2);
        float S = (M == NINF) ? 0.f
                              : (s1 * __expf(m1 - M) + s2 * __expf(m2 - M));
        sm[tid * H_ + h] = M;
        ss[tid * H_ + h] = S;
      }
    }
    __syncthreads();
  }
  if (tid < H_) {
    mmax [(b * N_ + j) * H_ + tid] = sm[tid];
    denom[(b * N_ + j) * H_ + tid] = ss[tid];
  }
}

// ---------------------------------------------------------------------------
// gs[b,j,h,d] = bf16( g[b,j,h,d] / denom[b,j,h] )  -- folds the softmax
// normalizer into the B matrix so the WMMA K-loop needs no rescale.
// ---------------------------------------------------------------------------
__global__ __launch_bounds__(256)
void scale_g_kernel(const float* __restrict__ g, const float* __restrict__ denom,
                    __bf16* __restrict__ gs) {
  int idx = blockIdx.x * 256 + threadIdx.x;
  if (idx >= BNF_) return;
  float den = denom[idx >> 5];               // idx/HID -> (b*N+n)*H+h
  float v = (den > 0.f) ? g[idx] / den : 0.f;
  gs[idx] = (__bf16)v;
}

// ---------------------------------------------------------------------------
// out[b,i,h,:] = sum_j P_ij * gs[b,j,h,:]
// P_ij = adj[i,j] ? exp(leakyrelu(src_i + dst_j) - m_j) : 0   (bf16, built in
// registers tile-by-tile; never materialized in memory).
// Block = (b, itile). 8 waves = 8 heads. Per K-tile: adj 16x32 staged via
// async-to-LDS (shared by all heads), dst/m staged per head by its own wave.
// Each wave owns a 16x32 output tile (two 16x16 WMMA accumulators reusing A).
// ---------------------------------------------------------------------------
__global__ __launch_bounds__(256)
void attn_wmma_kernel(const __bf16* __restrict__ gs,
                      const float* __restrict__ src, const float* __restrict__ dst,
                      const float* __restrict__ mmax, const int* __restrict__ adj,
                      float* __restrict__ out) {
  const int tid   = threadIdx.x;
  const int b     = blockIdx.x / (N_ / 16);
  const int itile = blockIdx.x % (N_ / 16);
  const int h     = tid >> 5;                // wave id == head
  const int lane  = tid & 31;
  const int half  = lane >> 4, l15 = lane & 15;
  const int i     = itile * 16 + l15;
  const float srci = src[(b * N_ + i) * H_ + h];

  int kmap[16];
#pragma unroll
  for (int s = 0; s < 16; s++) {
    int v = s >> 1, hb = s & 1;
    kmap[s] = (v < 4) ? (half * 8 + v * 2 + hb)
                      : (16 + half * 8 + (v - 4) * 2 + hb);
  }

  __shared__ int   adjT[16 * 32];            // 2 KB adj tile (all heads share)
  __shared__ float dstT[H_ * 32];            // per-head dst_j for this K-tile
  __shared__ float mT[H_ * 32];              // per-head m_j  for this K-tile
  v8f acc0 = {}, acc1 = {};

  // per-thread constants for the async adj staging (2 ints = 8B per thread)
  const int er = (tid * 2) >> 5, ec = (tid * 2) & 31;
  const unsigned adj_elem0 = (unsigned)((itile * 16 + er) * N_ + ec);
  const unsigned adjT_off  = lds_off(&adjT[tid * 2]);

  for (int k0 = 0; k0 < N_; k0 += 32) {
    __syncthreads();                         // protect LDS reuse across iters
    {
      unsigned boff = (adj_elem0 + (unsigned)k0) * 4u;
      asm volatile("global_load_async_to_lds_b64 %0, %1, %2"
                   :: "v"(adjT_off), "v"(boff), "s"(adj) : "memory");
    }
    // wave h stages its own head's 32 dst/m values (coalesced-by-wave)
    dstT[h * 32 + lane] = dst [(b * N_ + k0 + lane) * H_ + h];
    mT  [h * 32 + lane] = mmax[(b * N_ + k0 + lane) * H_ + h];
    if (k0 + 32 < N_)                        // speculative prefetch of next tile
      __builtin_prefetch(adj + (itile * 16) * N_ + k0 + 32, 0, 0);
    asm volatile("s_wait_asynccnt 0" ::: "memory");
    __syncthreads();

    // A fragment: P tile (16 rows i) x (32 cols j = k0..k0+31), bf16.
    v16bf a;
#pragma unroll
    for (int s = 0; s < 16; s++) {
      int kq = kmap[s];
      float p = 0.f;
      if (adjT[l15 * 32 + kq]) {
        float u = srci + dstT[h * 32 + kq];
        float e = (u >= 0.f) ? u : ALPHA * u;
        p = __expf(e - mT[h * 32 + kq]);
      }
      a[s] = (__bf16)p;
    }

    // B fragments: gs tile (32 rows j) x 16 cols d, for d-tiles 0 and 1.
    v16bf b0, b1;
#pragma unroll
    for (int s = 0; s < 16; s++) {
      int j = k0 + kmap[s];
      const __bf16* gp = gs + ((size_t)(b * N_ + j) * H_ + h) * HID_;
      b0[s] = gp[l15];
      b1[s] = gp[16 + l15];
    }
    acc0 = __builtin_amdgcn_wmma_f32_16x16x32_bf16(false, a, false, b0,
                                                   (short)0, acc0, false, false);
    acc1 = __builtin_amdgcn_wmma_f32_16x16x32_bf16(false, a, false, b1,
                                                   (short)0, acc1, false, false);
  }

#pragma unroll
  for (int v = 0; v < 8; v++) {
    int r = itile * 16 + v + half * 8;
    float* op = out + ((size_t)(b * N_ + r) * H_ + h) * HID_;
    op[l15]      = acc0[v];
    op[16 + l15] = acc1[v];
  }
}

// ---------------------------------------------------------------------------
// Training-mode BatchNorm1d over [B, N*F] (reduction over B=4 in registers),
// then ReLU, optional residual add.
// ---------------------------------------------------------------------------
__global__ __launch_bounds__(256)
void bn_relu_kernel(const float* __restrict__ y,
                    const float* __restrict__ gamma, const float* __restrict__ beta,
                    const float* __restrict__ residual, float* __restrict__ out) {
  int idx = blockIdx.x * 256 + threadIdx.x;
  if (idx >= NF_) return;
  float v0 = y[idx];
  float v1 = y[NF_ + idx];
  float v2 = y[2 * NF_ + idx];
  float v3 = y[3 * NF_ + idx];
  float mu = 0.25f * (v0 + v1 + v2 + v3);
  float d0 = v0 - mu, d1 = v1 - mu, d2 = v2 - mu, d3 = v3 - mu;
  float var = 0.25f * (d0 * d0 + d1 * d1 + d2 * d2 + d3 * d3);
  float rs = rsqrtf(var + EPS);
  float ga = gamma[idx], be = beta[idx];
  float o0 = fmaxf(ga * d0 * rs + be, 0.f);
  float o1 = fmaxf(ga * d1 * rs + be, 0.f);
  float o2 = fmaxf(ga * d2 * rs + be, 0.f);
  float o3 = fmaxf(ga * d3 * rs + be, 0.f);
  if (residual) {
    o0 += residual[idx];
    o1 += residual[NF_ + idx];
    o2 += residual[2 * NF_ + idx];
    o3 += residual[3 * NF_ + idx];
  }
  out[idx]           = o0;
  out[NF_ + idx]     = o1;
  out[2 * NF_ + idx] = o2;
  out[3 * NF_ + idx] = o3;
}

// ---------------------------------------------------------------------------
extern "C" void kernel_launch(void* const* d_in, const int* in_sizes, int n_in,
                              void* d_out, int out_size, void* d_ws, size_t ws_size,
                              hipStream_t stream) {
  const float* x   = (const float*)d_in[0];
  const int*   adj = (const int*)  d_in[1];
  const float* W1  = (const float*)d_in[2];
  const float* a1  = (const float*)d_in[3];
  const float* g1  = (const float*)d_in[4];
  const float* b1  = (const float*)d_in[5];
  const float* W2  = (const float*)d_in[6];
  const float* a2  = (const float*)d_in[7];
  const float* g2  = (const float*)d_in[8];
  const float* b2  = (const float*)d_in[9];
  float* outp = (float*)d_out;

  // Workspace layout (floats unless noted): ~14.5 MiB total.
  float*  g_buf = (float*)d_ws;                  // [BNF_]
  float*  att   = g_buf + BNF_;                  // [BNF_]
  float*  y1    = att + BNF_;                    // [BNF_]
  float*  srcb  = y1 + BNF_;                     // [BNH_]
  float*  dstb  = srcb + BNH_;                   // [BNH_]
  float*  mb    = dstb + BNH_;                   // [BNH_]
  float*  denb  = mb + BNH_;                     // [BNH_]
  __bf16* gs    = (__bf16*)(denb + BNH_);        // [BNF_] bf16

  const int gemm_blocks = (B_ * N_ / 16) * 2;              // 512
  const int attn_blocks = B_ * (N_ / 16);                  // 256

  // ---------------- Layer 1 ----------------
  gemm_wmma_kernel<<<gemm_blocks, 256, 0, stream>>>(x, W1, g_buf);
  src_dst_kernel<<<(BNH_ + 255) / 256, 256, 0, stream>>>(g_buf, a1, srcb, dstb);
  col_stats_kernel<<<B_ * N_, 256, 0, stream>>>(srcb, dstb, adj, mb, denb);
  scale_g_kernel<<<BNF_ / 256, 256, 0, stream>>>(g_buf, denb, gs);
  attn_wmma_kernel<<<attn_blocks, 256, 0, stream>>>(gs, srcb, dstb, mb, adj, att);
  bn_relu_kernel<<<NF_ / 256, 256, 0, stream>>>(att, g1, b1, nullptr, y1);

  // ---------------- Layer 2 ----------------
  gemm_wmma_kernel<<<gemm_blocks, 256, 0, stream>>>(y1, W2, g_buf);
  src_dst_kernel<<<(BNH_ + 255) / 256, 256, 0, stream>>>(g_buf, a2, srcb, dstb);
  col_stats_kernel<<<B_ * N_, 256, 0, stream>>>(srcb, dstb, adj, mb, denb);
  scale_g_kernel<<<BNF_ / 256, 256, 0, stream>>>(g_buf, denb, gs);
  attn_wmma_kernel<<<attn_blocks, 256, 0, stream>>>(gs, srcb, dstb, mb, adj, att);
  bn_relu_kernel<<<NF_ / 256, 256, 0, stream>>>(att, g2, b2, x, outp);

  (void)in_sizes; (void)n_in; (void)out_size; (void)ws_size;
}
